// DensityAndFeaturesMLP_22428319220068
// MI455X (gfx1250) — matrix operand order repaired and checked by reading, not compiled, not run on hardware
//
#include <hip/hip_runtime.h>

#define NGP_LEVELS 16
#define NGP_TABLE_SIZE (1u << 21)

typedef __attribute__((ext_vector_type(16))) _Float16 v16h;
typedef __attribute__((ext_vector_type(8)))  float    v8f;

// round(16 * (2048/16)^(i/15)) for i in 0..15; res <= 127 -> dense grid ((res+1)^3 <= 2^21)
__constant__ int kRes[NGP_LEVELS] = {16, 22, 31, 42, 58, 81, 111, 154,
                                     213, 294, 406, 562, 776, 1072, 1482, 2048};

__global__ __launch_bounds__(256, 1)
void ngp_mlp_kernel(const float* __restrict__ positions,  // [N,3]
                    const float* __restrict__ tables,     // [16, 2^21, 2]
                    const float* __restrict__ w0,         // [32,64]
                    const float* __restrict__ b0,         // [64]
                    const float* __restrict__ wd,         // [64,1]
                    const float* __restrict__ bd,         // [1]
                    const float* __restrict__ wf,         // [64,7]
                    const float* __restrict__ bf,         // [7]
                    float* __restrict__ out,              // [N*7] feats ++ [N] density
                    int N)
{
  __shared__ _Float16 enc[256][40];     // 256 points x 32 feats, padded rows
  __shared__ _Float16 hbuf[8][16][72];  // per-wave 16x64 hidden, padded rows

  const int t    = threadIdx.x;
  const int lane = t & 31;
  const int wave = t >> 5;
  const int hi   = (lane >= 16) ? 1 : 0;
  const int blk_base = blockIdx.x * 256;

  // ---------------- Phase 1: hash-grid encoding (one point per thread) ---------------
  {
    const int pt = blk_base + t;
    float px = 0.f, py = 0.f, pz = 0.f;
    if (pt < N) { px = positions[pt * 3 + 0]; py = positions[pt * 3 + 1]; pz = positions[pt * 3 + 2]; }
    #pragma unroll 1
    for (int l = 0; l < NGP_LEVELS; ++l) {
      const int res = kRes[l];
      const float* tab = tables + (size_t)l * (size_t)NGP_TABLE_SIZE * 2u;
      const float xf = px * (float)res, yf = py * (float)res, zf = pz * (float)res;
      const float fx = floorf(xf), fy = floorf(yf), fz = floorf(zf);
      const float wx = xf - fx, wy = yf - fy, wz = zf - fz;
      const float ux = 1.f - wx, uy = 1.f - wy, uz = 1.f - wz;
      const int ix = (int)fx, iy = (int)fy, iz = (int)fz;
      const bool dense = (res <= 127);
      const unsigned s = (unsigned)(res + 1);
      float f0 = 0.f, f1 = 0.f;
      #pragma unroll
      for (int c = 0; c < 8; ++c) {
        const unsigned cx = (unsigned)(ix + (c & 1));
        const unsigned cy = (unsigned)(iy + ((c >> 1) & 1));
        const unsigned cz = (unsigned)(iz + ((c >> 2) & 1));
        unsigned idx;
        if (dense) idx = cx + s * (cy + s * cz);
        else       idx = (cx ^ (cy * 2654435761u) ^ (cz * 805459861u)) & (NGP_TABLE_SIZE - 1u);
        const float wgt = ((c & 1) ? wx : ux) * (((c >> 1) & 1) ? wy : uy) * (((c >> 2) & 1) ? wz : uz);
        const float* e = tab + (size_t)idx * 2u;   // global_load_b64 gather
        f0 = fmaf(wgt, e[0], f0);
        f1 = fmaf(wgt, e[1], f1);
      }
      enc[t][2 * l + 0] = (_Float16)f0;
      enc[t][2 * l + 1] = (_Float16)f1;
    }
  }

  // ---------------- Weight fragments (registers, per wave) ----------------
  // B layout (KxN): lane&15 = column n; lanes<16 hold K=0..15, lanes>=16 hold K=16..31.
  v16h bw0[4];
  {
    const int kb = hi ? 16 : 0;
    #pragma unroll
    for (int c = 0; c < 4; ++c) {
      const int n = c * 16 + (lane & 15);
      #pragma unroll
      for (int j = 0; j < 16; ++j)
        bw0[c][j] = (_Float16)w0[(kb + j) * 64 + n];
    }
  }
  v16h bw2[2];  // [wf | wd | zero-pad] as 64x16, split into two K=32 chunks
  {
    const int n = lane & 15;
    #pragma unroll
    for (int kc = 0; kc < 2; ++kc) {
      const int kb = kc * 32 + (hi ? 16 : 0);
      #pragma unroll
      for (int j = 0; j < 16; ++j) {
        const int k = kb + j;
        float v = 0.f;
        if (n < 7)       v = wf[k * 7 + n];
        else if (n == 7) v = wd[k];
        bw2[kc][j] = (_Float16)v;
      }
    }
  }
  float bias1[4];
  #pragma unroll
  for (int c = 0; c < 4; ++c) bias1[c] = b0[c * 16 + (lane & 15)];
  float bias2;
  {
    const int n = lane & 15;
    bias2 = (n < 7) ? bf[n] : ((n == 7) ? bd[0] : 0.f);
  }

  __syncthreads();

  // ---------------- Phase 2: MLP via WMMA (each wave: 2 tiles of 16 points) ----------
  #pragma unroll 1
  for (int it = 0; it < 2; ++it) {
    const int tile = wave * 2 + it;
    const int row0 = tile * 16;

    // A fragment (16x32 f16): lane&15 = row m; lanes<16: K=0..15; lanes>=16: K=16..31
    v16h a;
    {
      const _Float16* r = &enc[row0 + (lane & 15)][0];
      const int k0 = hi ? 16 : 0;
      const int k1 = hi ? 24 : 8;
      #pragma unroll
      for (int j = 0; j < 8; ++j) { a[j] = r[k0 + j]; a[8 + j] = r[k1 + j]; }
    }

    // Layer 1: 16x32 @ 32x64 (+b0), 4 N-chunks of 16
    v8f acc[4];
    #pragma unroll
    for (int c = 0; c < 4; ++c) {
      #pragma unroll
      for (int g = 0; g < 8; ++g) acc[c][g] = bias1[c];
      acc[c] = __builtin_amdgcn_wmma_f32_16x16x32_f16(false, a, false, bw0[c],
                                                      (short)0, acc[c], false, false);
    }

    // ReLU + transpose h (16x64) through per-wave LDS buffer
    __syncthreads();
    {
      _Float16* hb = &hbuf[wave][0][0];
      const int n  = lane & 15;
      const int mb = hi ? 8 : 0;
      #pragma unroll
      for (int c = 0; c < 4; ++c) {
        #pragma unroll
        for (int g = 0; g < 8; ++g)
          hb[(mb + g) * 72 + c * 16 + n] = (_Float16)fmaxf(acc[c][g], 0.f);
      }
    }
    __syncthreads();

    // Layer 2: 16x64 @ 64x16(padded), K split into 2x32, bias preloaded
    v8f acc2;
    {
      #pragma unroll
      for (int g = 0; g < 8; ++g) acc2[g] = bias2;
      const _Float16* r = &hbuf[wave][lane & 15][0];
      #pragma unroll
      for (int kc = 0; kc < 2; ++kc) {
        v16h a2;
        const int k0 = kc * 32 + (hi ? 16 : 0);
        const int k1 = kc * 32 + (hi ? 24 : 8);
        #pragma unroll
        for (int j = 0; j < 8; ++j) { a2[j] = r[k0 + j]; a2[8 + j] = r[k1 + j]; }
        acc2 = __builtin_amdgcn_wmma_f32_16x16x32_f16(false, a2, false, bw2[kc],
                                                      (short)0, acc2, false, false);
      }
    }

    // Store from C/D layout: lane&15 = col n; VGPR g -> row g (lanes<16) / g+8 (lanes>=16)
    {
      const int n  = lane & 15;
      const int mb = hi ? 8 : 0;
      if (n < 8) {
        #pragma unroll
        for (int g = 0; g < 8; ++g) {
          const int p = blk_base + row0 + mb + g;
          if (p < N) {
            if (n < 7) out[(size_t)p * 7 + n]   = acc2[g];  // features [N,7]
            else       out[(size_t)N * 7 + p]   = acc2[g];  // density  [N,1]
          }
        }
      }
    }
  }
}

extern "C" void kernel_launch(void* const* d_in, const int* in_sizes, int n_in,
                              void* d_out, int out_size, void* d_ws, size_t ws_size,
                              hipStream_t stream) {
  const float* positions = (const float*)d_in[0];
  const float* tables    = (const float*)d_in[1];
  const float* w0        = (const float*)d_in[2];
  const float* b0        = (const float*)d_in[3];
  const float* wd        = (const float*)d_in[4];
  const float* bd        = (const float*)d_in[5];
  const float* wf        = (const float*)d_in[6];
  const float* bf        = (const float*)d_in[7];
  float* out = (float*)d_out;

  const int N = in_sizes[0] / 3;
  const int blocks = (N + 255) / 256;
  ngp_mlp_kernel<<<dim3(blocks), dim3(256), 0, stream>>>(
      positions, tables, w0, b0, wd, bd, wf, bf, out, N);
}
